// SlidingWindowAttention_61254823575773
// MI455X (gfx1250) — compile-verified, matrix operand order checked by reading, not run on hardware
//
#include <hip/hip_runtime.h>
#include <math.h>

// ---------- types ----------
typedef __bf16 bf16;
typedef __attribute__((ext_vector_type(8)))  __bf16 v8bf;
typedef __attribute__((ext_vector_type(16))) __bf16 v16bf;
typedef __attribute__((ext_vector_type(8)))  float  v8f;

__device__ __forceinline__ v16bf cat16(v8bf a, v8bf b) {
  return __builtin_shufflevector(a, b, 0,1,2,3,4,5,6,7,8,9,10,11,12,13,14,15);
}

__device__ __forceinline__ v8f wmma_bf16(v16bf a, v16bf b, v8f c) {
  // 8 args: (neg_a, A, neg_b, B, c_mod, C, reuse_a, reuse_b)
  return __builtin_amdgcn_wmma_f32_16x16x32_bf16(false, a, false, b, (short)0, c,
                                                 false, false);
}

// ---------- constants ----------
#define BB 2
#define SS 2048
#define HH 2048
#define NHD 16
#define HDD 128
#define WW 512

// ---------- tiled transpose + fp32->bf16 cast: out[n*K+k] = in[k*N+n] ----------
__global__ __launch_bounds__(256) void transpose_cast_kernel(
    const float* __restrict__ in, bf16* __restrict__ out, int K, int N) {
  __shared__ float tile[32][33];
  int k0 = blockIdx.x * 32;
  int n0 = blockIdx.y * 32;
  for (int i = 0; i < 4; ++i) {
    int idx = threadIdx.x + i * 256;
    int lk = idx >> 5, ln = idx & 31;
    tile[lk][ln] = in[(size_t)(k0 + lk) * N + n0 + ln];   // coalesced read
  }
  __syncthreads();
  for (int i = 0; i < 4; ++i) {
    int idx = threadIdx.x + i * 256;
    int ln = idx >> 5, lk = idx & 31;
    out[(size_t)(n0 + ln) * K + k0 + lk] = (bf16)tile[lk][ln];  // coalesced write
  }
}

// ---------- LayerNorm over H=2048 per row, write bf16 ----------
__global__ __launch_bounds__(256) void ln_kernel(const float* __restrict__ x,
                                                 const float* __restrict__ gamma,
                                                 const float* __restrict__ beta,
                                                 bf16* __restrict__ y) {
  int row = blockIdx.x;
  const float* xr = x + (size_t)row * HH;
  bf16* yr = y + (size_t)row * HH;
  float v[8];
  float s = 0.f, ss = 0.f;
  for (int i = 0; i < 8; ++i) {
    float t = xr[threadIdx.x + i * 256];
    v[i] = t; s += t; ss += t * t;
  }
  __shared__ float r1[256];
  __shared__ float r2[256];
  r1[threadIdx.x] = s; r2[threadIdx.x] = ss;
  __syncthreads();
  for (int off = 128; off > 0; off >>= 1) {
    if (threadIdx.x < off) {
      r1[threadIdx.x] += r1[threadIdx.x + off];
      r2[threadIdx.x] += r2[threadIdx.x + off];
    }
    __syncthreads();
  }
  float mu  = r1[0] * (1.0f / HH);
  float var = r2[0] * (1.0f / HH) - mu * mu;
  float inv = rsqrtf(var + 1e-6f);
  for (int i = 0; i < 8; ++i) {
    int c = threadIdx.x + i * 256;
    yr[c] = (bf16)(((v[i] - mu) * inv) * gamma[c] + beta[c]);
  }
}

// ---------- bf16 WMMA GEMM, barrier-free inner loop ----------
// C[M,N] = (A[M,K] x Bt^T + bias) * alpha, with Bt pre-transposed [N][K].
// Wave computes a 32x64 strip: 2 A-frags x 4 B-frags -> 8 WMMAs per K-step,
// every B fragment reused across two A fragments (12 b128 loads : 8 WMMAs).
// split==1: bf16 out, [B,NH,S,HD]; split==2: bf16 out, [B,NH,HD,S];
// split==0: fp32 row-major (Cf) or bf16 row-major (Cb).
__global__ __launch_bounds__(256) void gemm_bf16_kernel(
    const bf16* __restrict__ A, const bf16* __restrict__ Bt,
    int M, int N, int K, float alpha, const float* __restrict__ bias,
    float* __restrict__ Cf, bf16* __restrict__ Cb, int split) {
  int lane  = threadIdx.x & 31;
  int wave  = threadIdx.x >> 5;
  int col16 = lane & 15;
  int kb    = (lane >> 4) * 8;   // K offset of this lane half (A/B frag)
  int hi8   = kb;                // row offset of this lane half (C frag)
  int tileM = blockIdx.y * 256 + wave * 32;
  int tileN = blockIdx.x * 64;

  const bf16* arow0 = A + (size_t)(tileM + col16) * K;
  const bf16* arow1 = A + (size_t)(tileM + 16 + col16) * K;
  const bf16* brow0 = Bt + (size_t)(tileN + 0  + col16) * K;
  const bf16* brow1 = Bt + (size_t)(tileN + 16 + col16) * K;
  const bf16* brow2 = Bt + (size_t)(tileN + 32 + col16) * K;
  const bf16* brow3 = Bt + (size_t)(tileN + 48 + col16) * K;

  v8f acc00 = {}, acc01 = {}, acc02 = {}, acc03 = {};
  v8f acc10 = {}, acc11 = {}, acc12 = {}, acc13 = {};
  for (int k0 = 0; k0 < K; k0 += 32) {
    __builtin_prefetch(arow0 + k0 + 512, 0, 1);
    __builtin_prefetch(brow0 + k0 + 512, 0, 1);
    v16bf a0 = cat16(*(const v8bf*)(arow0 + k0 + kb),
                     *(const v8bf*)(arow0 + k0 + 16 + kb));
    v16bf a1 = cat16(*(const v8bf*)(arow1 + k0 + kb),
                     *(const v8bf*)(arow1 + k0 + 16 + kb));
    v16bf b0 = cat16(*(const v8bf*)(brow0 + k0 + kb),
                     *(const v8bf*)(brow0 + k0 + 16 + kb));
    v16bf b1 = cat16(*(const v8bf*)(brow1 + k0 + kb),
                     *(const v8bf*)(brow1 + k0 + 16 + kb));
    v16bf b2 = cat16(*(const v8bf*)(brow2 + k0 + kb),
                     *(const v8bf*)(brow2 + k0 + 16 + kb));
    v16bf b3 = cat16(*(const v8bf*)(brow3 + k0 + kb),
                     *(const v8bf*)(brow3 + k0 + 16 + kb));
    acc00 = wmma_bf16(a0, b0, acc00);
    acc01 = wmma_bf16(a0, b1, acc01);
    acc02 = wmma_bf16(a0, b2, acc02);
    acc03 = wmma_bf16(a0, b3, acc03);
    acc10 = wmma_bf16(a1, b0, acc10);
    acc11 = wmma_bf16(a1, b1, acc11);
    acc12 = wmma_bf16(a1, b2, acc12);
    acc13 = wmma_bf16(a1, b3, acc13);
  }

  v8f accs[2][4] = {{acc00, acc01, acc02, acc03}, {acc10, acc11, acc12, acc13}};
  for (int h = 0; h < 2; ++h) {
    for (int t = 0; t < 4; ++t) {
      for (int r = 0; r < 8; ++r) {
        int row = tileM + h * 16 + r + hi8;
        int col = tileN + t * 16 + col16;
        float val = accs[h][t][r];
        if (bias) val += bias[col];
        val *= alpha;
        if (split == 1) {          // [B*S, NH*HD] -> [B,NH,S,HD]
          int b = row / SS, s = row - b * SS;
          int n = col / HDD, hd = col - n * HDD;
          Cb[(((size_t)(b * NHD + n)) * SS + s) * HDD + hd] = (bf16)val;
        } else if (split == 2) {   // [B*S, NH*HD] -> [B,NH,HD,S] (transposed V)
          int b = row / SS, s = row - b * SS;
          int n = col / HDD, hd = col - n * HDD;
          Cb[(((size_t)(b * NHD + n)) * HDD + hd) * SS + s] = (bf16)val;
        } else if (Cf) {
          Cf[(size_t)row * N + col] = val;
        } else {
          Cb[(size_t)row * N + col] = (bf16)val;
        }
      }
    }
  }
}

// ---------- flash attention (inverted-window mask, per reference) ----------
// Q/K: [B,NH,S,HD] bf16 (Q pre-scaled by 1/sqrt(HD));
// Vt:  [B,NH,HD,S] bf16 (transposed so B-frags are contiguous b128 loads);
// O:   [B*S, NH*HD] bf16.
__global__ __launch_bounds__(32) void flash_kernel(const bf16* __restrict__ Q,
                                                   const bf16* __restrict__ Kv,
                                                   const bf16* __restrict__ Vt,
                                                   bf16* __restrict__ O) {
  __shared__ __align__(16) bf16 plds[16 * 32];
  int lane  = threadIdx.x;
  int col16 = lane & 15;
  int hib   = lane >> 4;
  int kb    = hib * 8;
  int i0 = blockIdx.x * 16;
  int n  = blockIdx.y;
  int b  = blockIdx.z;
  size_t base = ((size_t)(b * NHD + n)) * SS * HDD;  // same base for Q,K,Vt

  v16bf qf[4];
  for (int c = 0; c < 4; ++c) {
    const bf16* p = Q + base + (size_t)(i0 + col16) * HDD + c * 32 + kb;
    qf[c] = cat16(*(const v8bf*)p, *(const v8bf*)(p + 16));
  }

  float m[8], l[8], alr[8];
  v8f acc[8];
  const v8f vzero = {};
  for (int r = 0; r < 8; ++r) { m[r] = -3.0e38f; l[r] = 0.f; }
  for (int nc = 0; nc < 8; ++nc) acc[nc] = vzero;

  for (int j0 = 0; j0 < SS; j0 += 32) {
    // valid iff (j > i) || (j <= i - W): reference keeps OUT-of-window scores
    bool any_valid = (j0 + 31 > i0) || (j0 <= i0 + 15 - WW);
    if (!any_valid) continue;

    if (j0 + 32 < SS) {
      __builtin_prefetch(Kv + base + (size_t)(j0 + 32 + col16) * HDD, 0, 1);
      __builtin_prefetch(Vt + base + (size_t)col16 * SS + j0 + 32, 0, 1);
    }

    v8f s0 = vzero, s1 = vzero;
    for (int c = 0; c < 4; ++c) {
      const bf16* p0 = Kv + base + (size_t)(j0 + col16) * HDD + c * 32 + kb;
      const bf16* p1 = Kv + base + (size_t)(j0 + 16 + col16) * HDD + c * 32 + kb;
      v16bf kf0 = cat16(*(const v8bf*)p0, *(const v8bf*)(p0 + 16));
      v16bf kf1 = cat16(*(const v8bf*)p1, *(const v8bf*)(p1 + 16));
      s0 = wmma_bf16(qf[c], kf0, s0);
      s1 = wmma_bf16(qf[c], kf1, s1);
    }

    for (int r = 0; r < 8; ++r) {
      int i  = i0 + r + hib * 8;
      int jA = j0 + col16;
      int jB = jA + 16;
      float sa = ((jA > i) || (jA <= i - WW)) ? s0[r] : -3.0e38f;
      float sb = ((jB > i) || (jB <= i - WW)) ? s1[r] : -3.0e38f;
      float t = fmaxf(sa, sb);
      t = fmaxf(t, __shfl_xor(t, 1, 16));
      t = fmaxf(t, __shfl_xor(t, 2, 16));
      t = fmaxf(t, __shfl_xor(t, 4, 16));
      t = fmaxf(t, __shfl_xor(t, 8, 16));
      float mn = fmaxf(m[r], t);
      float al = __expf(m[r] - mn);
      float p0 = __expf(sa - mn);
      float p1 = __expf(sb - mn);
      float rs = p0 + p1;
      rs += __shfl_xor(rs, 1, 16);
      rs += __shfl_xor(rs, 2, 16);
      rs += __shfl_xor(rs, 4, 16);
      rs += __shfl_xor(rs, 8, 16);
      m[r] = mn;
      l[r] = l[r] * al + rs;
      alr[r] = al;
      s0[r] = p0;
      s1[r] = p1;
    }
    for (int nc = 0; nc < 8; ++nc)
      for (int r = 0; r < 8; ++r) acc[nc][r] *= alr[r];

    // C-layout P -> A-layout via LDS (16x32 bf16, single-wave WG)
    for (int r = 0; r < 8; ++r) {
      int row = r + hib * 8;
      plds[row * 32 + col16]      = (bf16)s0[r];
      plds[row * 32 + 16 + col16] = (bf16)s1[r];
    }
    __syncthreads();
    v16bf pf = cat16(*(const v8bf*)&plds[col16 * 32 + kb],
                     *(const v8bf*)&plds[col16 * 32 + 16 + kb]);
    for (int nc = 0; nc < 8; ++nc) {
      const bf16* pv = Vt + base + (size_t)(nc * 16 + col16) * SS + j0 + kb;
      v16bf vf = cat16(*(const v8bf*)pv, *(const v8bf*)(pv + 16));
      acc[nc] = wmma_bf16(pf, vf, acc[nc]);
    }
    __syncthreads();
  }

  float inv[8];
  for (int r = 0; r < 8; ++r) inv[r] = 1.0f / l[r];
  for (int nc = 0; nc < 8; ++nc) {
    for (int r = 0; r < 8; ++r) {
      size_t row = (size_t)b * SS + i0 + r + hib * 8;
      int col = n * HDD + nc * 16 + col16;
      O[row * (NHD * HDD) + col] = (bf16)(acc[nc][r] * inv[r]);
    }
  }
}

// ---------- host launcher ----------
extern "C" void kernel_launch(void* const* d_in, const int* in_sizes, int n_in,
                              void* d_out, int out_size, void* d_ws, size_t ws_size,
                              hipStream_t stream) {
  (void)in_sizes; (void)n_in; (void)out_size; (void)ws_size;
  const float* x   = (const float*)d_in[0];
  const float* lns = (const float*)d_in[1];
  const float* lnb = (const float*)d_in[2];
  const float* wq  = (const float*)d_in[3];
  const float* bq  = (const float*)d_in[4];
  const float* wk  = (const float*)d_in[5];
  const float* bk  = (const float*)d_in[6];
  const float* wv  = (const float*)d_in[7];
  const float* bv  = (const float*)d_in[8];
  const float* wo  = (const float*)d_in[9];
  const float* bo  = (const float*)d_in[10];
  float* out = (float*)d_out;

  const size_t M  = (size_t)BB * SS;          // 4096
  const size_t WN = (size_t)HH * NHD * HDD;   // 2048*2048

  char* ws = (char*)d_ws;
  bf16* y_bf  = (bf16*)ws; ws += M * HH * 2;
  bf16* wqt   = (bf16*)ws; ws += WN * 2;      // transposed [N][K]
  bf16* wkt   = (bf16*)ws; ws += WN * 2;
  bf16* wvt   = (bf16*)ws; ws += WN * 2;
  bf16* wot   = (bf16*)ws; ws += WN * 2;
  bf16* q_ws  = (bf16*)ws; ws += M * HH * 2;  // [B,NH,S,HD]
  bf16* k_ws  = (bf16*)ws; ws += M * HH * 2;  // [B,NH,S,HD]
  bf16* vt_ws = (bf16*)ws; ws += M * HH * 2;  // [B,NH,HD,S]
  bf16* o_ws  = (bf16*)ws; ws += M * HH * 2;  // [B*S, NH*HD]   (~112 MB total)

  // one-time weight transpose+cast: W[K=H][N] -> Wt[N][K]
  dim3 tg(HH / 32, HH / 32);
  transpose_cast_kernel<<<tg, 256, 0, stream>>>(wq, wqt, HH, HH);
  transpose_cast_kernel<<<tg, 256, 0, stream>>>(wk, wkt, HH, HH);
  transpose_cast_kernel<<<tg, 256, 0, stream>>>(wv, wvt, HH, HH);
  transpose_cast_kernel<<<tg, 256, 0, stream>>>(wo, wot, HH, HH);

  // LayerNorm
  ln_kernel<<<(int)M, 256, 0, stream>>>(x, lns, lnb, y_bf);

  // QKV projections; q = (y*Wq + bq)/sqrt(HD) per reference ordering
  dim3 gg(HH / 64, (int)M / 256);
  const float qscale = 0.08838834764831845f;  // 1/sqrt(128)
  gemm_bf16_kernel<<<gg, 256, 0, stream>>>(y_bf, wqt, (int)M, HH, HH, qscale, bq,
                                           nullptr, q_ws, 1);
  gemm_bf16_kernel<<<gg, 256, 0, stream>>>(y_bf, wkt, (int)M, HH, HH, 1.0f, bk,
                                           nullptr, k_ws, 1);
  gemm_bf16_kernel<<<gg, 256, 0, stream>>>(y_bf, wvt, (int)M, HH, HH, 1.0f, bv,
                                           nullptr, vt_ws, 2);

  // streaming-softmax attention
  flash_kernel<<<dim3(SS / 16, NHD, BB), 32, 0, stream>>>(q_ws, k_ws, vt_ws, o_ws);

  // output projection + bias -> fp32 d_out
  gemm_bf16_kernel<<<gg, 256, 0, stream>>>(o_ws, wot, (int)M, HH, HH, 1.0f, bo,
                                           out, nullptr, 0);
}
